// Nets_71554155151856
// MI455X (gfx1250) — matrix operand order repaired and checked by reading, not compiled, not run on hardware
//
#include <hip/hip_runtime.h>
#include <hip/hip_bf16.h>

typedef _Float16 v16h __attribute__((ext_vector_type(16)));
typedef _Float16 v8h  __attribute__((ext_vector_type(8)));
typedef float    v8f  __attribute__((ext_vector_type(8)));

#define NN 8192
#define EE 65536
#define DD 480
#define SS 128
#define LL 64
#define HH 4

__device__ __forceinline__ float siluf(float x) { return x / (1.f + __expf(-x)); }

__device__ __forceinline__ v8f wmma16(v16h a, v16h b, v8f c) {
  // D = A(16x32 f16) * B(32x16 f16) + C(16x16 f32)
  return __builtin_amdgcn_wmma_f32_16x16x32_f16(false, a, false, b, (short)0, c, false, false);
}

// CDNA5 async global->LDS copy (ASYNCcnt-tracked), 16 bytes per lane.
__device__ __forceinline__ void async_b128_to_lds(unsigned lds_off, const void* gaddr) {
  asm volatile("global_load_async_to_lds_b128 %0, %1, off"
               :: "v"(lds_off), "v"((unsigned long long)(size_t)gaddr)
               : "memory");
}
__device__ __forceinline__ void wait_async_le1() {
  asm volatile("s_wait_asynccnt 0x1" ::: "memory");
}
__device__ __forceinline__ void wait_async_le0() {
  asm volatile("s_wait_asynccnt 0x0" ::: "memory");
}

// ---------------------------------------------------------------------------
// Generic WMMA GEMM:  C[M x 480] = A[M x K] @ W[K x 480]
// Wt is the weight pre-transposed to [480 x K] f16 (B fragment = 2 runs of 8
// halves per lane).  Block: 320 threads = 10 wave32, covering 32 rows x 480
// cols (wave w owns cols 16w, 16w+160, 16w+320; 2 M-tiles each -> 6 WMMA/k).
// A tiles (2 x 16x32) are double-buffered in LDS via async global->LDS DMA.
// AMODE: 0 = direct f16 buffer (stride K)
//        1 = concat gather [node_n[src] | node_n[dst] | edge_n]  (K = 1440)
//        2 = v_h scaled at fragment read by alpha[row][k/120]    (K = 480)
// EPI:   0 = store f16
//        1 = silu(acc + bias[col]) -> f16
//        2 = silu(acc) * aux[idx]  -> f16
//        3 = store f32
//        4 = f32 +=
// ---------------------------------------------------------------------------
template<int AMODE, int EPI>
__global__ __launch_bounds__(320)
void gemm_wmma(const _Float16* __restrict__ A, int K,
               const _Float16* __restrict__ Wt,
               const int* __restrict__ gsrc, const int* __restrict__ gdst,
               const _Float16* __restrict__ nodeN, const _Float16* __restrict__ edgeN,
               const float* __restrict__ alpha,
               const float* __restrict__ bias,
               const _Float16* __restrict__ aux,
               _Float16* __restrict__ outH, float* __restrict__ outF)
{
  __shared__ __align__(16) _Float16 At[2][2][16 * 32];   // [buf][mtile][row*32+k]
  const int tid  = threadIdx.x;
  const int lane = tid & 31;
  const int w    = tid >> 5;            // 0..9
  const int rowBase = blockIdx.x << 5;  // 32 rows per block

  v8f acc[2][3] = {};

  const int arow = lane & 15;           // A row within tile
  const int kbA  = (lane >> 4) << 3;    // 0 or 8   (A k sub-offset)
  const int kbB  = (lane >> 4) << 3;    // 0 or 8   (B k sub-offset)
  const int bcol = lane & 15;           // B column within tile

  // Loader role: waves 0..3 (128 lanes), one 16-byte chunk per lane.
  const bool loader = (tid < 128);
  const int lr  = tid >> 2;             // 0..31 : row within block
  const int lch = tid & 3;              // 0..3  : 8-half chunk
  const unsigned ldsBase =
      (unsigned)(size_t)(&At[0][lr >> 4][(lr & 15) * 32 + lch * 8]);
  const unsigned ldsStride = (unsigned)(sizeof(_Float16) * 2 * 16 * 32); // buf stride

  auto issue = [&](int kt, int buf) {
    if (!loader) return;
    const int k = (kt << 5) + (lch << 3);
    const int row = rowBase + lr;
    const _Float16* p;
    if (AMODE == 1) {
      if (k < DD)          p = nodeN + (size_t)gsrc[row] * DD + k;
      else if (k < 2 * DD) p = nodeN + (size_t)gdst[row] * DD + (k - DD);
      else                 p = edgeN + (size_t)row * DD + (k - 2 * DD);
    } else if (AMODE == 2) {
      p = A + (size_t)row * DD + k;
    } else {
      p = A + (size_t)row * K + k;
    }
    async_b128_to_lds(ldsBase + (unsigned)buf * ldsStride, p);
  };

  const int nk = K >> 5;
  issue(0, 0);

  for (int i = 0; i < nk; ++i) {
    const int buf = i & 1;
    if (i + 1 < nk) { issue(i + 1, buf ^ 1); wait_async_le1(); }
    else            { wait_async_le0(); }
    __syncthreads();

    v16h av[2];
    #pragma unroll
    for (int t = 0; t < 2; ++t) {
      v8h alo = *(const v8h*)(&At[buf][t][arow * 32 + kbA]);
      v8h ahi = *(const v8h*)(&At[buf][t][arow * 32 + 16 + kbA]);
      if (AMODE == 2) {
        const int row = rowBase + t * 16 + arow;
        const int k0 = i << 5;
        const _Float16 s0 = (_Float16)alpha[row * HH + (k0 + kbA) / 120];
        const _Float16 s1 = (_Float16)alpha[row * HH + (k0 + 16 + kbA) / 120];
        alo = alo * s0;
        ahi = ahi * s1;
      }
      av[t] = __builtin_shufflevector(alo, ahi,
          0, 1, 2, 3, 4, 5, 6, 7, 8, 9, 10, 11, 12, 13, 14, 15);
    }
    #pragma unroll
    for (int j = 0; j < 3; ++j) {
      const int col = w * 16 + j * 160 + bcol;
      const v8h blo = *(const v8h*)(Wt + (size_t)col * K + (i << 5) + kbB);
      const v8h bhi = *(const v8h*)(Wt + (size_t)col * K + (i << 5) + 16 + kbB);
      const v16h bv = __builtin_shufflevector(blo, bhi,
          0, 1, 2, 3, 4, 5, 6, 7, 8, 9, 10, 11, 12, 13, 14, 15);
      acc[0][j] = wmma16(av[0], bv, acc[0][j]);
      acc[1][j] = wmma16(av[1], bv, acc[1][j]);
    }
    __syncthreads();
  }

  const int n  = lane & 15;
  const int mo = (lane >> 4) << 3;
  #pragma unroll
  for (int t = 0; t < 2; ++t) {
    #pragma unroll
    for (int j = 0; j < 3; ++j) {
      const int col = w * 16 + j * 160 + n;
      #pragma unroll
      for (int r = 0; r < 8; ++r) {
        const size_t idx = (size_t)(rowBase + t * 16 + mo + r) * DD + col;
        const float x = acc[t][j][r];
        if (EPI == 0)      outH[idx] = (_Float16)x;
        else if (EPI == 1) { const float y = x + bias[col]; outH[idx] = (_Float16)siluf(y); }
        else if (EPI == 2) { outH[idx] = (_Float16)(siluf(x) * (float)aux[idx]); }
        else if (EPI == 3) outF[idx] = x;
        else               outF[idx] += x;
      }
    }
  }
}

// ---------------------------------------------------------------------------
// RMS-normalize rows of x (cols = 480) -> f16, plus raw f16 copy.
// One wave32 per row, 8 rows per block.
// ---------------------------------------------------------------------------
__global__ __launch_bounds__(256)
void rms_convert(const float* __restrict__ x, _Float16* __restrict__ xn,
                 _Float16* __restrict__ xraw, int rows)
{
  const int wid = threadIdx.x >> 5, lane = threadIdx.x & 31;
  const int row = blockIdx.x * 8 + wid;
  if (row >= rows) return;
  const float* xr = x + (size_t)row * DD;
  float ss = 0.f;
  for (int c = lane; c < DD; c += 32) { const float v = xr[c]; ss += v * v; }
  #pragma unroll
  for (int m = 16; m > 0; m >>= 1) ss += __shfl_xor(ss, m, 32);
  const float sc = rsqrtf(ss * (1.f / DD) + 1e-6f);
  for (int c = lane; c < DD; c += 32) {
    const float v = xr[c];
    xn[(size_t)row * DD + c]   = (_Float16)(v * sc);
    xraw[(size_t)row * DD + c] = (_Float16)v;
  }
}

__global__ void wtrans(const float* __restrict__ W, _Float16* __restrict__ Wt,
                       int K, int Ncols)
{
  const int idx = blockIdx.x * 256 + threadIdx.x;
  if (idx >= K * Ncols) return;
  const int k = idx / Ncols, n = idx - k * Ncols;
  Wt[(size_t)n * K + k] = (_Float16)W[idx];
}

__global__ void f2h(const float* __restrict__ x, _Float16* __restrict__ y, int n)
{
  const int i = blockIdx.x * 256 + threadIdx.x;
  if (i < n) y[i] = (_Float16)x[i];
}

// ---------------------------------------------------------------------------
// Per-edge small MLPs: q/k heads + length-embedding attention bias -> alpha.
// 64 threads per edge, 4 edges per 256-thread block. LayerNorm via LDS trees.
// ---------------------------------------------------------------------------
__device__ __forceinline__ float groupsum64(float v, float* red, int t)
{
  red[t] = v;
  __syncthreads();
  #pragma unroll
  for (int s = 32; s > 0; s >>= 1) {
    if (t < s) red[t] += red[t + s];
    __syncthreads();
  }
  const float r = red[0];
  __syncthreads();
  return r;
}

__global__ __launch_bounds__(256)
void edge_mlp(const _Float16* __restrict__ nodeN, const float* __restrict__ ele,
              const int* __restrict__ esrc, const int* __restrict__ edst,
              const float* qW1, const float* qb1, const float* qg1, const float* qc1,
              const float* qW2, const float* qb2,
              const float* kW1, const float* kb1, const float* kg1, const float* kc1,
              const float* kW2, const float* kb2,
              const float* aW1, const float* ab1, const float* ag1, const float* ac1,
              const float* aW2, const float* ab2, const float* ag2, const float* ac2,
              const float* aW3, const float* ab3,
              float* __restrict__ alpha)
{
  __shared__ float s_in[4][128];
  __shared__ float s_hid[4][64];
  __shared__ float s_vec[4][64];
  __shared__ float s_red[4][64];
  __shared__ float s_qk[4][4];
  const int tid = threadIdx.x;
  const int g = tid >> 6;
  const int t = tid & 63;
  const int e = blockIdx.x * 4 + g;
  float* red = s_red[g];

  // ---- Q path (scal[dst]) ----
  {
    const int nd = edst[e];
    s_in[g][t]      = (float)nodeN[(size_t)nd * DD + t];
    s_in[g][t + 64] = (float)nodeN[(size_t)nd * DD + t + 64];
  }
  __syncthreads();
  float y = qb1[t];
  for (int j = 0; j < SS; ++j) y += s_in[g][j] * qW1[j * 64 + t];
  float sum = groupsum64(y, red, t);
  float sq  = groupsum64(y * y, red, t);
  float mu = sum * (1.f / 64.f);
  float var = sq * (1.f / 64.f) - mu * mu;
  float h = siluf((y - mu) * rsqrtf(var + 1e-6f) * qg1[t] + qc1[t]);
  s_hid[g][t] = h;
  __syncthreads();
  float qv = qb2[t];
  for (int j = 0; j < 64; ++j) qv += s_hid[g][j] * qW2[j * 64 + t];
  s_vec[g][t] = qv;
  __syncthreads();

  // ---- K path (scal[src]) ----
  {
    const int ns = esrc[e];
    s_in[g][t]      = (float)nodeN[(size_t)ns * DD + t];
    s_in[g][t + 64] = (float)nodeN[(size_t)ns * DD + t + 64];
  }
  __syncthreads();
  y = kb1[t];
  for (int j = 0; j < SS; ++j) y += s_in[g][j] * kW1[j * 64 + t];
  sum = groupsum64(y, red, t);
  sq  = groupsum64(y * y, red, t);
  mu = sum * (1.f / 64.f);
  var = sq * (1.f / 64.f) - mu * mu;
  h = siluf((y - mu) * rsqrtf(var + 1e-6f) * kg1[t] + kc1[t]);
  s_hid[g][t] = h;
  __syncthreads();
  float kv = kb2[t];
  for (int j = 0; j < 64; ++j) kv += s_hid[g][j] * kW2[j * 64 + t];
  s_vec[g][t] = s_vec[g][t] * kv;   // per-thread element, no race
  __syncthreads();
  if (t < HH) {
    float a = 0.f;
    for (int d = 0; d < 16; ++d) a += s_vec[g][t * 16 + d];
    s_qk[g][t] = a * 0.25f;         // 1/sqrt(DK)
  }
  __syncthreads();

  // ---- attention-bias path on length embedding ----
  s_in[g][t] = ele[(size_t)e * LL + t];
  __syncthreads();
  y = ab1[t];
  for (int j = 0; j < 64; ++j) y += s_in[g][j] * aW1[j * 64 + t];
  sum = groupsum64(y, red, t); sq = groupsum64(y * y, red, t);
  mu = sum * (1.f / 64.f); var = sq * (1.f / 64.f) - mu * mu;
  h = siluf((y - mu) * rsqrtf(var + 1e-6f) * ag1[t] + ac1[t]);
  s_hid[g][t] = h;
  __syncthreads();
  y = ab2[t];
  for (int j = 0; j < 64; ++j) y += s_hid[g][j] * aW2[j * 64 + t];
  sum = groupsum64(y, red, t); sq = groupsum64(y * y, red, t);
  mu = sum * (1.f / 64.f); var = sq * (1.f / 64.f) - mu * mu;
  h = siluf((y - mu) * rsqrtf(var + 1e-6f) * ag2[t] + ac2[t]);
  s_in[g][t] = h;
  __syncthreads();
  if (t < HH) {
    float a3 = ab3[t];
    for (int j = 0; j < 64; ++j) a3 += s_in[g][j] * aW3[j * HH + t];
    alpha[(size_t)e * HH + t] = s_qk[g][t] + a3;
  }
}

// ---------------------------------------------------------------------------
// Segment softmax over edge_dst (order-preserving uint encoding for f32 max).
// ---------------------------------------------------------------------------
__device__ __forceinline__ unsigned fenc(float f) {
  const unsigned u = __float_as_uint(f);
  return (u & 0x80000000u) ? ~u : (u | 0x80000000u);
}
__device__ __forceinline__ float fdec(unsigned e) {
  const unsigned u = (e & 0x80000000u) ? (e ^ 0x80000000u) : ~e;
  return __uint_as_float(u);
}

__global__ void seg_max(const float* __restrict__ alpha, const int* __restrict__ edst,
                        unsigned* __restrict__ amax)
{
  const int i = blockIdx.x * 256 + threadIdx.x;   // over E*H
  const int e = i >> 2, hh = i & 3;
  atomicMax(&amax[edst[e] * HH + hh], fenc(alpha[i]));
}
__global__ void seg_exp(const float* __restrict__ alpha, const int* __restrict__ edst,
                        const unsigned* __restrict__ amax,
                        float* __restrict__ ex, float* __restrict__ denom)
{
  const int i = blockIdx.x * 256 + threadIdx.x;
  const int e = i >> 2, hh = i & 3;
  const float m = fdec(amax[edst[e] * HH + hh]);
  const float v = __expf(alpha[i] - m);
  ex[i] = v;
  atomicAdd(&denom[edst[e] * HH + hh], v);
}
__global__ void seg_norm(const float* __restrict__ denom, const int* __restrict__ edst,
                         float* __restrict__ ex)
{
  const int i = blockIdx.x * 256 + threadIdx.x;
  const int e = i >> 2, hh = i & 3;
  ex[i] = ex[i] / (denom[edst[e] * HH + hh] + 1e-12f);
}

// v * asoft scatter-add into node_msg
__global__ void scatter_add(const _Float16* __restrict__ v_h, const float* __restrict__ asoft,
                            const int* __restrict__ edst, float* __restrict__ node_msg)
{
  const int i = blockIdx.x * 256 + threadIdx.x;   // over E*480
  const int e = i / DD, c = i - e * DD;
  const float val = (float)v_h[i] * asoft[e * HH + c / 120];
  atomicAdd(node_msg + (size_t)edst[e] * DD + c, val);
}

// onsite edges (|edge_vec| < 1e-10): v <- msg @ o_W   (rare, predicated)
__global__ __launch_bounds__(64)
void onsite_fix(const float* __restrict__ evec, const _Float16* __restrict__ msg_h,
                const float* __restrict__ oW, _Float16* __restrict__ v_h)
{
  __shared__ float sm[DD];
  __shared__ int flag;
  const int e = blockIdx.x;
  if (threadIdx.x == 0) {
    const float a = evec[e * 3], b = evec[e * 3 + 1], c = evec[e * 3 + 2];
    flag = (a * a + b * b + c * c) < 1e-20f;
  }
  __syncthreads();
  if (!flag) return;
  for (int j = threadIdx.x; j < DD; j += 64) sm[j] = (float)msg_h[(size_t)e * DD + j];
  __syncthreads();
  for (int c = threadIdx.x; c < DD; c += 64) {
    float acc = 0.f;
    for (int j = 0; j < DD; ++j) acc += sm[j] * oW[j * DD + c];
    v_h[(size_t)e * DD + c] = (_Float16)acc;
  }
}

// ---------------------------------------------------------------------------
extern "C" void kernel_launch(void* const* d_in, const int* in_sizes, int n_in,
                              void* d_out, int out_size, void* d_ws, size_t ws_size,
                              hipStream_t stream)
{
  (void)in_sizes; (void)n_in; (void)out_size; (void)ws_size;
  const float* node_fea = (const float*)d_in[0];
  const float* edge_fea = (const float*)d_in[1];
  const float* ele      = (const float*)d_in[2];
  const float* evec     = (const float*)d_in[3];
  const int*   esrc     = (const int*)d_in[4];
  const int*   edst     = (const int*)d_in[5];
  const float* qW1 = (const float*)d_in[6],  *qb1 = (const float*)d_in[7];
  const float* qg1 = (const float*)d_in[8],  *qc1 = (const float*)d_in[9];
  const float* qW2 = (const float*)d_in[10], *qb2 = (const float*)d_in[11];
  const float* kW1 = (const float*)d_in[12], *kb1 = (const float*)d_in[13];
  const float* kg1 = (const float*)d_in[14], *kc1 = (const float*)d_in[15];
  const float* kW2 = (const float*)d_in[16], *kb2 = (const float*)d_in[17];
  const float* aW1 = (const float*)d_in[18], *ab1 = (const float*)d_in[19];
  const float* ag1 = (const float*)d_in[20], *ac1 = (const float*)d_in[21];
  const float* aW2 = (const float*)d_in[22], *ab2 = (const float*)d_in[23];
  const float* ag2 = (const float*)d_in[24], *ac2 = (const float*)d_in[25];
  const float* aW3 = (const float*)d_in[26], *ab3 = (const float*)d_in[27];
  const float* pre_W = (const float*)d_in[28];
  const float* f_W  = (const float*)d_in[29], *f_b = (const float*)d_in[30];
  const float* v_W1 = (const float*)d_in[31], *v_W2 = (const float*)d_in[32];
  const float* o_W  = (const float*)d_in[33], *l_W  = (const float*)d_in[34];
  const float* le_W = (const float*)d_in[35], *ns_W = (const float*)d_in[36];
  const float* es_W = (const float*)d_in[37];

  char* wp = (char*)d_ws;
  auto carve = [&](size_t bytes) -> void* {
    void* r = (void*)wp; wp += (bytes + 255) & ~(size_t)255; return r;
  };
  _Float16* node_n_h   = (_Float16*)carve((size_t)NN * DD * 2);
  _Float16* node_raw_h = (_Float16*)carve((size_t)NN * DD * 2);
  _Float16* edge_n_h   = (_Float16*)carve((size_t)EE * DD * 2);
  _Float16* edge_raw_h = (_Float16*)carve((size_t)EE * DD * 2);
  _Float16* ele_h      = (_Float16*)carve((size_t)EE * LL * 2);
  _Float16* filt_h     = (_Float16*)carve((size_t)EE * DD * 2);
  _Float16* msg_h      = (_Float16*)carve((size_t)EE * DD * 2);
  _Float16* u_h        = (_Float16*)carve((size_t)EE * DD * 2);
  _Float16* v_h        = (_Float16*)carve((size_t)EE * DD * 2);
  float*    node_msg   = (float*)carve((size_t)NN * DD * 4);
  _Float16* node_msg_h = (_Float16*)carve((size_t)NN * DD * 2);
  float*    alpha      = (float*)carve((size_t)EE * HH * 4);
  float*    asoft      = (float*)carve((size_t)EE * HH * 4);
  unsigned* amax       = (unsigned*)carve((size_t)NN * HH * 4);
  float*    denom      = (float*)carve((size_t)NN * HH * 4);
  _Float16* pre_Wt = (_Float16*)carve((size_t)3 * DD * DD * 2);
  _Float16* f_Wt   = (_Float16*)carve((size_t)LL * DD * 2);
  _Float16* vW1t   = (_Float16*)carve((size_t)DD * DD * 2);
  _Float16* vW2t   = (_Float16*)carve((size_t)DD * DD * 2);
  _Float16* l_Wt   = (_Float16*)carve((size_t)DD * DD * 2);
  _Float16* le_Wt  = (_Float16*)carve((size_t)DD * DD * 2);
  _Float16* ns_Wt  = (_Float16*)carve((size_t)DD * DD * 2);
  _Float16* es_Wt  = (_Float16*)carve((size_t)DD * DD * 2);

  hipMemsetAsync(node_msg, 0, (size_t)NN * DD * 4, stream);
  hipMemsetAsync(amax,     0, (size_t)NN * HH * 4, stream);
  hipMemsetAsync(denom,    0, (size_t)NN * HH * 4, stream);

  auto WT = [&](const float* W, _Float16* Wt, int K, int Ncols) {
    const int total = K * Ncols;
    wtrans<<<(total + 255) / 256, 256, 0, stream>>>(W, Wt, K, Ncols);
  };
  WT(pre_W, pre_Wt, 3 * DD, DD);
  WT(f_W,  f_Wt,  LL, DD);
  WT(v_W1, vW1t, DD, DD);
  WT(v_W2, vW2t, DD, DD);
  WT(l_W,  l_Wt, DD, DD);
  WT(le_W, le_Wt, DD, DD);
  WT(ns_W, ns_Wt, DD, DD);
  WT(es_W, es_Wt, DD, DD);

  rms_convert<<<NN / 8, 256, 0, stream>>>(node_fea, node_n_h, node_raw_h, NN);
  rms_convert<<<EE / 8, 256, 0, stream>>>(edge_fea, edge_n_h, edge_raw_h, EE);
  f2h<<<(EE * LL + 255) / 256, 256, 0, stream>>>(ele, ele_h, EE * LL);

  edge_mlp<<<EE / 4, 256, 0, stream>>>(node_n_h, ele, esrc, edst,
      qW1, qb1, qg1, qc1, qW2, qb2,
      kW1, kb1, kg1, kc1, kW2, kb2,
      aW1, ab1, ag1, ac1, aW2, ab2, ag2, ac2, aW3, ab3, alpha);

  seg_max<<<EE * HH / 256, 256, 0, stream>>>(alpha, edst, amax);
  seg_exp<<<EE * HH / 256, 256, 0, stream>>>(alpha, edst, amax, asoft, denom);
  seg_norm<<<EE * HH / 256, 256, 0, stream>>>(denom, edst, asoft);

  // filt = silu(ele @ f_W + f_b)
  gemm_wmma<0, 1><<<EE / 32, 320, 0, stream>>>(ele_h, LL, f_Wt,
      nullptr, nullptr, nullptr, nullptr, nullptr, f_b, nullptr, filt_h, nullptr);
  // msg = [node_n[src] | node_n[dst] | edge_n] @ pre_W
  gemm_wmma<1, 0><<<EE / 32, 320, 0, stream>>>(nullptr, 3 * DD, pre_Wt,
      esrc, edst, node_n_h, edge_n_h, nullptr, nullptr, nullptr, msg_h, nullptr);
  // u = silu(msg @ v_W1) * filt
  gemm_wmma<0, 2><<<EE / 32, 320, 0, stream>>>(msg_h, DD, vW1t,
      nullptr, nullptr, nullptr, nullptr, nullptr, nullptr, filt_h, u_h, nullptr);
  // v = u @ v_W2
  gemm_wmma<0, 0><<<EE / 32, 320, 0, stream>>>(u_h, DD, vW2t,
      nullptr, nullptr, nullptr, nullptr, nullptr, nullptr, nullptr, v_h, nullptr);
  // onsite replacement v <- msg @ o_W
  onsite_fix<<<EE, 64, 0, stream>>>(evec, msg_h, o_W, v_h);
  // node_msg = segment_sum(v * asoft)
  scatter_add<<<EE * DD / 256, 256, 0, stream>>>(v_h, asoft, edst, node_msg);
  f2h<<<(NN * DD + 255) / 256, 256, 0, stream>>>(node_msg, node_msg_h, NN * DD);

  float* node_out = (float*)d_out;
  float* edge_out = (float*)d_out + (size_t)NN * DD;
  // edge_out = (v * alpha) @ le_W + edge_res @ es_W
  gemm_wmma<2, 3><<<EE / 32, 320, 0, stream>>>(v_h, DD, le_Wt,
      nullptr, nullptr, nullptr, nullptr, alpha, nullptr, nullptr, nullptr, edge_out);
  gemm_wmma<0, 4><<<EE / 32, 320, 0, stream>>>(edge_raw_h, DD, es_Wt,
      nullptr, nullptr, nullptr, nullptr, nullptr, nullptr, nullptr, nullptr, edge_out);
  // node_out = node_msg @ l_W + node_res @ ns_W
  gemm_wmma<0, 3><<<NN / 32, 320, 0, stream>>>(node_msg_h, DD, l_Wt,
      nullptr, nullptr, nullptr, nullptr, nullptr, nullptr, nullptr, nullptr, node_out);
  gemm_wmma<0, 4><<<NN / 32, 320, 0, stream>>>(node_raw_h, DD, ns_Wt,
      nullptr, nullptr, nullptr, nullptr, nullptr, nullptr, nullptr, nullptr, node_out);
}
